// AdaptiveConv2DMod_5025111736437
// MI455X (gfx1250) — matrix operand order repaired and checked by reading, not compiled, not run on hardware
//
#include <hip/hip_runtime.h>
#include <hip/hip_bf16.h>

// ---------------------------------------------------------------------------
// AdaptiveConv2DMod for MI455X (gfx1250), wave32 + WMMA f16 implicit GEMM
// with async global->LDS double-buffered staging (ASYNCcnt).
// B=16, C=256, H=W=64, O=256, N=4 kernels, 3x3, pad=1.
//
// Kernel 1: fmap f32 [b][i][h][x] -> f16 channel-last [b][h][x][i]  (ws)
// Kernel 2: per-(b,o) modulated+demodulated f16 weight rows, K = kl*256+i (ws)
// Kernel 3: per-(b,h) implicit GEMM, 16 waves x (16 o x 64 px), K=2304 in
//           four 64-channel passes, async-filled LDS slabs, double buffered.
// ---------------------------------------------------------------------------

typedef __attribute__((ext_vector_type(16))) _Float16 v16h;
typedef __attribute__((ext_vector_type(8)))  _Float16 v8h;
typedef __attribute__((ext_vector_type(8)))  float    v8f;

struct HalfPair { v8h lo, hi; };
__device__ __forceinline__ v16h make_frag(v8h lo, v8h hi) {
    HalfPair p{lo, hi};
    return __builtin_bit_cast(v16h, p);
}

// Low 32 bits of the flat address of an LDS object == LDS byte address
// (ISA 10.2: aperture in addr[63:32], LDS offset in addr[31:0]).
__device__ __forceinline__ unsigned lds_addr32(const void* p) {
    return (unsigned)(unsigned long long)p;
}

// ASYNCcnt-tracked 16B global->LDS copy (per lane), CDNA5 async path.
__device__ __forceinline__ void async_copy_b128(unsigned lds_byte, const void* gaddr) {
    asm volatile("global_load_async_to_lds_b128 %0, %1, off"
                 :: "v"(lds_byte), "v"(gaddr) : "memory");
}

#define B_    16
#define C_    256
#define O_    256
#define HW_   64
#define KL_   9
#define KDIM  2304          // 9 * 256
#define IP2   72            // LDS i-pitch per 64-channel pass (mult of 8)
#define SLAB  (3 * 66 * IP2) // halfs per buffer (28,512 B)

// ---------------------------------------------------------------------------
// Kernel 1: fmap -> f16 channel-last.  grid (64 h, 16 b), 256 threads.
// ---------------------------------------------------------------------------
__global__ __launch_bounds__(256)
void fmap_to_f16_kernel(const float* __restrict__ fmap,     // [16][256][64][64]
                        _Float16*    __restrict__ out16)    // [16][64][64][256]
{
    __shared__ _Float16 tile[64 * 264];    // [x][i], padded pitch
    const int h = blockIdx.x, b = blockIdx.y;
    const int tid = threadIdx.x;
    const int x = tid & 63, g = tid >> 6;

    for (int i = g; i < C_; i += 4) {      // coalesced in x
        float v = fmap[(((size_t)(b * C_ + i)) * HW_ + h) * HW_ + x];
        tile[x * 264 + i] = (_Float16)v;
    }
    __syncthreads();

    _Float16* ob = out16 + ((size_t)(b * HW_ + h)) * HW_ * C_;
#pragma unroll
    for (int k = 0; k < 8; ++k) {          // 2048 16B chunks, coalesced b128
        int j  = tid + k * 256;
        int xx = j >> 5;
        int ip = (j & 31) * 8;
        *(v8h*)&ob[xx * C_ + ip] = *(const v8h*)&tile[xx * 264 + ip];
    }
}

// ---------------------------------------------------------------------------
// Kernel 2: weight generation.  grid = B*O (b innermost), 256 threads = i.
// ---------------------------------------------------------------------------
__global__ __launch_bounds__(256)
void wmod_gen_kernel(const float* __restrict__ weights,   // [4][256][256][9]
                     const float* __restrict__ mod,       // [16][256]
                     const float* __restrict__ kmod,      // [16][4]
                     _Float16*    __restrict__ A)         // [16][256][2304]
{
    const int blk = blockIdx.x;
    const int b   = blk & (B_ - 1);
    const int o   = blk >> 4;
    const int i   = threadIdx.x;

    float l0 = kmod[b * 4 + 0], l1 = kmod[b * 4 + 1];
    float l2 = kmod[b * 4 + 2], l3 = kmod[b * 4 + 3];
    float mx = fmaxf(fmaxf(l0, l1), fmaxf(l2, l3));
    float e0 = __expf(l0 - mx), e1 = __expf(l1 - mx);
    float e2 = __expf(l2 - mx), e3 = __expf(l3 - mx);
    float inv = 1.0f / (e0 + e1 + e2 + e3);
    e0 *= inv; e1 *= inv; e2 *= inv; e3 *= inv;

    const float m = mod[b * C_ + i] + 1.0f;
    const size_t nstride = (size_t)O_ * C_ * KL_;
    const float* wp = weights + ((size_t)o * C_ + i) * KL_;

    float v[KL_];
    float ss = 0.0f;
#pragma unroll
    for (int kl = 0; kl < KL_; ++kl) {
        float w = e0 * wp[kl] + e1 * wp[nstride + kl]
                + e2 * wp[2 * nstride + kl] + e3 * wp[3 * nstride + kl];
        w *= m;
        v[kl] = w;
        ss += w * w;
    }

    __shared__ float red[256];
    red[i] = ss;
    __syncthreads();
#pragma unroll
    for (int s = 128; s > 0; s >>= 1) {
        if (i < s) red[i] += red[i + s];
        __syncthreads();
    }
    const float scale = rsqrtf(fmaxf(red[0], 1e-8f));

    _Float16* arow = A + (size_t)(b * O_ + o) * KDIM;
#pragma unroll
    for (int kl = 0; kl < KL_; ++kl)
        arow[kl * C_ + i] = (_Float16)(v[kl] * scale);
}

// ---------------------------------------------------------------------------
// Kernel 3: implicit-GEMM conv with WMMA + async double-buffered LDS.
// grid (64 h, 16 b), 512 threads = 16 waves.
// ---------------------------------------------------------------------------
__global__ __launch_bounds__(512)
void adaptive_conv_wmma_kernel(const _Float16* __restrict__ fmap16, // [16][64][64][256]
                               const _Float16* __restrict__ A,      // [16][256][2304]
                               float*          __restrict__ out)    // [16][256][64][64]
{
    __shared__ _Float16 sm[2][SLAB];       // 2 x 28.5 KB double buffer
    __shared__ _Float16 dump[8];           // sink for out-of-range async copies

    const int h    = blockIdx.x;
    const int b    = blockIdx.y;
    const int tid  = threadIdx.x;
    const int lane = tid & 31;
    const int wv   = tid >> 5;
    const int obase = wv * 16;
    const int lh   = lane & 15;
    const int lhi  = lane >> 4;

    // zero pad columns c=0 and c=65 in both buffers (never overwritten)
    for (int j = tid; j < 2 * 3 * 2 * IP2; j += 512) {
        int buf = j / (3 * 2 * IP2);
        int q   = j - buf * (3 * 2 * IP2);
        int r   = q / (2 * IP2);
        int q2  = q - r * 2 * IP2;
        int c   = (q2 < IP2) ? 0 : 65;
        int ii  = (q2 < IP2) ? q2 : q2 - IP2;
        sm[buf][(r * 66 + c) * IP2 + ii] = (_Float16)0.0f;
    }

    const unsigned smBase  = lds_addr32(&sm[0][0]);
    const unsigned dumpOff = lds_addr32(&dump[0]);

    // async fill of one 64-channel pass: exactly 3 async ops per thread
    // (uniform ASYNCcnt), out-of-range rows -> dump + ds-stored zeros.
    auto fill_async = [&](int ci, int buf) {
#pragma unroll
        for (int k = 0; k < 3; ++k) {
            int j   = tid + k * 512;        // 1536 16B chunks
            int r   = j >> 9;               // fmap row h-1..h+1
            int rem = j & 511;
            int xx  = rem >> 3;             // pixel 0..63
            int p   = rem & 7;              // 8-half sub-chunk
            int hr  = h + r - 1;
            bool ok = ((unsigned)hr < (unsigned)HW_);
            const _Float16* g = fmap16 +
                (((size_t)((b * HW_ + (ok ? hr : h)) * HW_ + xx)) * C_ + ci + p * 8);
            unsigned real = smBase +
                (unsigned)((buf * SLAB + (r * 66 + xx + 1) * IP2 + p * 8) * 2);
            async_copy_b128(ok ? real : dumpOff, g);
            if (!ok) {
                v8h z = {};
                *(v8h*)&sm[buf][(r * 66 + xx + 1) * IP2 + p * 8] = z;
            }
        }
    };

    v8f acc[4] = {};
    const _Float16* arow = A + (size_t)(b * O_ + obase + lh) * KDIM + lhi * 8;

    fill_async(0,  0);                      // prologue: passes 0 and 1 in flight
    fill_async(64, 1);

    for (int ps = 0; ps < 4; ++ps) {
        // retire this pass's copies (in-order; next pass's 3 may stay pending)
        if (ps < 3) asm volatile("s_wait_asynccnt 0x3" ::: "memory");
        else        asm volatile("s_wait_asynccnt 0x0" ::: "memory");
        __syncthreads();

        const int buf = ps & 1;
        const int ci  = ps * 64;
        const _Float16* smb = &sm[buf][0];

        for (int dh = 0; dh < 3; ++dh) {
            for (int dw = 0; dw < 3; ++dw) {
#pragma unroll
                for (int i0 = 0; i0 < 64; i0 += 32) {
                    // A frag: lanes 0-15 K k0..+7/+16..23, lanes 16-31 +8/+24
                    const _Float16* ap = arow + ((dh * 3 + dw) * C_ + ci + i0);
                    const v16h afrag =
                        make_frag(*(const v8h*)ap, *(const v8h*)(ap + 16));
#pragma unroll
                    for (int t = 0; t < 4; ++t) {
                        // B frag: lanes 0-15 K k0..k0+7/+16..23, lanes 16-31 +8/+24
                        const _Float16* bp = smb +
                            (size_t)(dh * 66 + dw + t * 16 + lh) * IP2 + i0 + lhi * 8;
                        const v16h bfrag =
                            make_frag(*(const v8h*)bp, *(const v8h*)(bp + 16));
                        acc[t] = __builtin_amdgcn_wmma_f32_16x16x32_f16(
                            false, afrag, false, bfrag,
                            (short)0, acc[t], false, false);
                    }
                }
            }
        }
        __syncthreads();                    // buffer free before re-fill
        if (ps + 2 < 4) fill_async((ps + 2) * 64, buf);
    }

    // C/D layout: element r -> lanes 0-15 M=r, lanes 16-31 M=r+8
    const int mo = lhi * 8;
#pragma unroll
    for (int t = 0; t < 4; ++t) {
#pragma unroll
        for (int r = 0; r < 8; ++r) {
            out[(((size_t)(b * O_ + obase + mo + r) * HW_) + h) * HW_
                + t * 16 + lh] = acc[t][r];
        }
    }
}

// ---------------------------------------------------------------------------
extern "C" void kernel_launch(void* const* d_in, const int* in_sizes, int n_in,
                              void* d_out, int out_size, void* d_ws, size_t ws_size,
                              hipStream_t stream) {
    const float* fmap    = (const float*)d_in[0];  // [16][256][64][64]
    const float* mod     = (const float*)d_in[1];  // [16][256]
    const float* kmod    = (const float*)d_in[2];  // [16][4]
    const float* weights = (const float*)d_in[3];  // [4][256][256][3][3]
    float*       out     = (float*)d_out;          // [16][256][64][64]

    // ws layout: [A: 16*256*2304 f16 = 18.9 MB][fmap16: 16*64*64*256 f16 = 33.6 MB]
    _Float16* A      = (_Float16*)d_ws;
    _Float16* fmap16 = (_Float16*)((char*)d_ws + (size_t)B_ * O_ * KDIM * sizeof(_Float16));

    fmap_to_f16_kernel<<<dim3(HW_, B_), dim3(256), 0, stream>>>(fmap, fmap16);
    wmod_gen_kernel<<<dim3(B_ * O_), dim3(256), 0, stream>>>(weights, mod, kmod, A);
    adaptive_conv_wmma_kernel<<<dim3(HW_, B_), dim3(512), 0, stream>>>(fmap16, A, out);
}
